// AttentionModule_51110110822821
// MI455X (gfx1250) — matrix operand order repaired and checked by reading, not compiled, not run on hardware
//
#include <hip/hip_runtime.h>

// ---------------------------------------------------------------------------
// Self-attention (B=8, C=256, N=4096, D=32), flash-attention style, bf16 WMMA
// + Tensor Data Mover (TDM) double-buffered K/V staging into padded LDS.
//   Phase 0: x[b][c][n] f32 -> xT[b][n][c] bf16 (LDS tile transpose); weights.
//   Phase 1: q/k transposed -> qT/kT[b][n][d]; v -> v[b][c][n]. All fragment
//            loads pair-contiguous b128; stores coalesced.
//   Phase 2: 8 waves/block, 16 query rows each. Key chunks of 32 DMA'd by
//            tensor_load_to_lds (pad 16B per 64B row -> conflict-free stride),
//            double buffered on TENSORcnt, one barrier per chunk. Online
//            softmax, O over 16 c-tiles, transposed b128 epilogue.
// ---------------------------------------------------------------------------

#define BATCH 8
#define CCH   256
#define NN    4096
#define DD    32

typedef __attribute__((ext_vector_type(8)))  float        v8f;
typedef __attribute__((ext_vector_type(4)))  float        v4f;
typedef __attribute__((ext_vector_type(16))) __bf16       v16bf;
typedef __attribute__((ext_vector_type(4)))  unsigned int u32x4;
typedef __attribute__((ext_vector_type(8)))  int          i32x8;
typedef __attribute__((ext_vector_type(4)))  int          i32x4;

#if defined(__AMDGCN__) && __has_builtin(__builtin_amdgcn_tensor_load_to_lds)
#define USE_TDM 1
#else
#define USE_TDM 0
#endif

union BFfrag {
    v16bf v;
    unsigned short h[16];
    unsigned int   u[8];
};

__device__ __forceinline__ unsigned short f2bf(float f) {
    unsigned int u = __float_as_uint(f);
    unsigned int r = u + 0x7FFFu + ((u >> 16) & 1u);   // round-to-nearest-even
    return (unsigned short)(r >> 16);
}

// ---------------- Phase 0: conversions ------------------------------------

__global__ __launch_bounds__(256) void k_cvt_xT(const float* __restrict__ x,
                                                unsigned short* __restrict__ xT) {
    __shared__ float tile[32][33];
    const int n0 = blockIdx.x * 32, c0 = blockIdx.y * 32, b = blockIdx.z;
    const int tx = threadIdx.x & 31, ty = threadIdx.x >> 5;
#pragma unroll
    for (int rep = 0; rep < 4; ++rep) {
        int c_l = ty + rep * 8;
        tile[c_l][tx] = x[(size_t)(b * CCH + c0 + c_l) * NN + n0 + tx];
    }
    __syncthreads();
#pragma unroll
    for (int rep = 0; rep < 4; ++rep) {
        int n_l = ty + rep * 8;
        xT[(size_t)(b * NN + n0 + n_l) * CCH + c0 + tx] = f2bf(tile[tx][n_l]);
    }
}

__global__ __launch_bounds__(256) void k_cvt_w(const float* __restrict__ Wq, const float* __restrict__ bq,
                                               const float* __restrict__ Wk, const float* __restrict__ bk,
                                               const float* __restrict__ Wv, const float* __restrict__ bv,
                                               unsigned short* __restrict__ Wall,
                                               float* __restrict__ ball) {
    int i = blockIdx.x * blockDim.x + threadIdx.x;
    if (i < 320 * 256) {
        int row = i >> 8, c = i & 255;
        float w = (row < 32) ? Wq[row * 256 + c]
                : (row < 64) ? Wk[(row - 32) * 256 + c]
                             : Wv[(row - 64) * 256 + c];
        Wall[i] = f2bf(w);
    }
    if (i < 320)
        ball[i] = (i < 32) ? bq[i] : (i < 64) ? bk[i - 32] : bv[i - 64];
}

// ---------------- Phase 1a: q/k projection, transposed output -------------

__global__ __launch_bounds__(32) void k_projQK(const unsigned short* __restrict__ xT,
                                               const unsigned short* __restrict__ Wall,
                                               const float* __restrict__ ball,
                                               unsigned short* __restrict__ qT,
                                               unsigned short* __restrict__ kT) {
    const int nt = blockIdx.x, ot = blockIdx.y, b = blockIdx.z;
    const int lane = threadIdx.x & 31, lrow = lane & 15, hi = lane >> 4;
    const int og = ot * 16;
    const unsigned short* xrow = xT + (size_t)(b * NN + nt * 16 + lrow) * CCH;
    const unsigned short* wrow = Wall + (size_t)(og + lrow) * 256;
    const float ballv = ball[og + lrow];

    v8f acc = {};
#pragma unroll
    for (int cc = 0; cc < 8; ++cc) {
        BFfrag a, bw;
#pragma unroll
        for (int j = 0; j < 8; ++j) {
            int c0 = cc * 32 + hi * 16 + 2 * j;
            a.u[j] = *(const unsigned int*)(xrow + c0);
        }
#pragma unroll
        for (int j = 0; j < 8; ++j) {
            int c0 = cc * 32 + hi * 16 + 2 * j;
            bw.u[j] = *(const unsigned int*)(wrow + c0);
        }
        acc = __builtin_amdgcn_wmma_f32_16x16x32_bf16(false, a.v, false, bw.v,
                                                      (short)0, acc, false, false);
    }
    unsigned short* dst = (ot < 2) ? qT : kT;
    const int obase = (ot < 2) ? og : og - 32;
#pragma unroll
    for (int r = 0; r < 8; ++r) {
        int n = nt * 16 + r + hi * 8;
        dst[(size_t)(b * NN + n) * DD + obase + lrow] = f2bf(acc[r] + ballv);
    }
}

// ---------------- Phase 1b: v projection, natural output ------------------

__global__ __launch_bounds__(32) void k_projV(const unsigned short* __restrict__ xT,
                                              const unsigned short* __restrict__ Wall,
                                              const float* __restrict__ ball,
                                              unsigned short* __restrict__ vb) {
    const int nt = blockIdx.x, ot = blockIdx.y, b = blockIdx.z;
    const int lane = threadIdx.x & 31, lrow = lane & 15, hi = lane >> 4;
    const unsigned short* wrow = Wall + (size_t)(64 + ot * 16 + lrow) * 256;
    const unsigned short* xrow = xT + (size_t)(b * NN + nt * 16 + lrow) * CCH;

    v8f acc = {};
#pragma unroll
    for (int cc = 0; cc < 8; ++cc) {
        BFfrag a, bx;
#pragma unroll
        for (int j = 0; j < 8; ++j) {
            int c0 = cc * 32 + hi * 16 + 2 * j;
            a.u[j] = *(const unsigned int*)(wrow + c0);
        }
#pragma unroll
        for (int j = 0; j < 8; ++j) {
            int c0 = cc * 32 + hi * 16 + 2 * j;
            bx.u[j] = *(const unsigned int*)(xrow + c0);
        }
        acc = __builtin_amdgcn_wmma_f32_16x16x32_bf16(false, a.v, false, bx.v,
                                                      (short)0, acc, false, false);
    }
#pragma unroll
    for (int r = 0; r < 8; ++r) {
        int ch = ot * 16 + r + hi * 8;
        int n  = nt * 16 + lrow;
        vb[(size_t)(b * CCH + ch) * NN + n] = f2bf(acc[r] + ball[64 + ch]);
    }
}

// ---------------- Phase 2: streaming attention ----------------------------

#define LDSPAD 40   // halfword row stride (80 B): TDM pad 16B per 64B row

#if USE_TDM
// TDM D# group0: count=1, lds byte addr, 57-bit global byte addr, type=2.
__device__ __forceinline__ u32x4 tdm_g0(unsigned ldsOff, unsigned long long g) {
    u32x4 g0;
    g0.x = 1u;
    g0.y = ldsOff;
    g0.z = (unsigned)g;
    g0.w = (unsigned)(g >> 32) | (2u << 30);
    return g0;
}
__device__ __forceinline__ void tdm_issue(unsigned ldsOff, const void* gptr, i32x8 g1) {
    i32x4 z4 = {};
    i32x8 z8 = {};
    __builtin_amdgcn_tensor_load_to_lds(tdm_g0(ldsOff, (unsigned long long)(uintptr_t)gptr),
                                        g1, z4, z4, z8, 0);
}
#endif

__global__ __launch_bounds__(256) void k_attn(const unsigned short* __restrict__ qT,
                                              const unsigned short* __restrict__ kT,
                                              const unsigned short* __restrict__ vb,
                                              const float* __restrict__ x,
                                              float* __restrict__ out) {
    __shared__ unsigned short ldsK[2][32 * LDSPAD];           // [m][d], d contig
    __shared__ unsigned short ldsV[2][CCH * LDSPAD];          // [c][m], m contig
    __shared__ __align__(16) unsigned char ldsScr[8 * 1280];  // per-wave P / O-transpose

    const int t    = threadIdx.x;
    const int wv   = t >> 5;
    const int lane = t & 31;
    const int lrow = lane & 15;
    const int hi   = lane >> 4;
    const int b     = blockIdx.y;
    const int qbase = blockIdx.x * 128 + wv * 16;

    const unsigned short* kbb = kT + (size_t)b * NN * DD;
    const unsigned short* vbb = vb + (size_t)b * CCH * NN;
    unsigned short* myP  = (unsigned short*)(ldsScr + wv * 1280);  // [16 row][40]
    float*          scrF = (float*)(ldsScr + wv * 1280);           // [16 c][20]

    // Q A-fragment (16n x 32d), pairs contiguous in qT[n][d].
    BFfrag qa;
    {
        const unsigned short* qrow = qT + (size_t)(b * NN + qbase + lrow) * DD;
#pragma unroll
        for (int j = 0; j < 8; ++j) {
            int d0 = hi * 16 + 2 * j;
            qa.u[j] = *(const unsigned int*)(qrow + d0);
        }
    }

    v8f O[16];
#pragma unroll
    for (int ct = 0; ct < 16; ++ct) O[ct] = {};
    float rowmax[8], rowsum[8];
#pragma unroll
    for (int r = 0; r < 8; ++r) { rowmax[r] = -1e30f; rowsum[r] = 0.0f; }

#if USE_TDM
    // D# group1 constants: data_size=2B, pad_enable, pad every 16 DW by 4 DW.
    //   dw0 = (1<<16)|(1<<20)|(3<<22)|(3<<25) = 0x06D10000
    const i32x8 g1V = { 0x06D10000, 0x10000000, 0x01000000, 0x00200000,
                        256, 4096, 0, 0 };   // dims: 4096x256 tensor, 32x256 tile
    const i32x8 g1K = { 0x06D10000, 0x00200000, 0x00200000, 0x00200000,
                        32, 32, 0, 0 };      // dims: 32x32 tensor, 32x32 tile
    if (wv == 0) {                           // prologue: DMA chunk 0 -> buffer 0
        tdm_issue((unsigned)(uintptr_t)&ldsK[0][0], kbb, g1K);
        tdm_issue((unsigned)(uintptr_t)&ldsV[0][0], vbb, g1V);
    }
#endif

#pragma unroll 1
    for (int ch = 0; ch < NN / 32; ++ch) {
        const int buf = ch & 1;
        const int m0  = ch * 32;
#if USE_TDM
        if (wv == 0)
            asm volatile("s_wait_tensorcnt 0x0" ::: "memory");
        __syncthreads();   // chunk ch visible to all; prior buf reads finished
        if (wv == 0 && ch + 1 < NN / 32) {   // overlap: DMA chunk ch+1
            const int nm0 = m0 + 32;
            tdm_issue((unsigned)(uintptr_t)&ldsK[buf ^ 1][0],
                      kbb + (size_t)nm0 * DD, g1K);
            tdm_issue((unsigned)(uintptr_t)&ldsV[buf ^ 1][0],
                      vbb + nm0, g1V);
        }
#else
        __syncthreads();
        {   // K chunk: 32 rows of kT[m][32d] -> ldsK[m][d]; one uint2/thread.
            int m = t >> 3, seg = t & 7;
            *(uint2*)(&ldsK[buf][0] + m * LDSPAD + seg * 4) =
                *(const uint2*)(kbb + (size_t)(m0 + m) * DD + seg * 4);
        }
#pragma unroll
        for (int rep = 0; rep < 8; ++rep) {  // V chunk: 256c x 32m
            int idx = rep * 256 + t;
            int c = idx >> 3, mg = idx & 7;
            *(uint2*)(&ldsV[buf][0] + c * LDSPAD + mg * 4) =
                *(const uint2*)(vbb + (size_t)c * NN + m0 + mg * 4);
        }
        __syncthreads();
#endif

        // S = Q^T K : two 16x16 tiles covering 32 keys (K-dim = D = 32).
        BFfrag bk0, bk1;
#pragma unroll
        for (int j = 0; j < 8; ++j) {
            int d0 = hi * 16 + 2 * j;
            bk0.u[j] = *(const unsigned int*)(&ldsK[buf][0] + lrow * LDSPAD + d0);
            bk1.u[j] = *(const unsigned int*)(&ldsK[buf][0] + (16 + lrow) * LDSPAD + d0);
        }
        v8f S0 = {}, S1 = {};
        S0 = __builtin_amdgcn_wmma_f32_16x16x32_bf16(false, qa.v, false, bk0.v,
                                                     (short)0, S0, false, false);
        S1 = __builtin_amdgcn_wmma_f32_16x16x32_bf16(false, qa.v, false, bk1.v,
                                                     (short)0, S1, false, false);

        // Online softmax: each row lives across 16 lanes of a half-wave.
        v8f corrv;
#pragma unroll
        for (int r = 0; r < 8; ++r) {
            float cmax = fmaxf(S0[r], S1[r]);
#pragma unroll
            for (int msk = 1; msk <= 8; msk <<= 1)
                cmax = fmaxf(cmax, __shfl_xor(cmax, msk, 32));
            float nm = fmaxf(rowmax[r], cmax);
            float cr = __expf(rowmax[r] - nm);
            rowmax[r] = nm;
            float p0 = __expf(S0[r] - nm);
            float p1 = __expf(S1[r] - nm);
            S0[r] = p0; S1[r] = p1;
            float ps = p0 + p1;
#pragma unroll
            for (int msk = 1; msk <= 8; msk <<= 1)
                ps += __shfl_xor(ps, msk, 32);
            rowsum[r] = rowsum[r] * cr + ps;
            corrv[r] = cr;
        }
#pragma unroll
        for (int ct = 0; ct < 16; ++ct) O[ct] *= corrv;

        // P (C-layout f32) -> A-layout bf16 via wave-private LDS transpose.
#pragma unroll
        for (int r = 0; r < 8; ++r) {
            int row = r + hi * 8;
            myP[row * LDSPAD + lrow]      = f2bf(S0[r]);
            myP[row * LDSPAD + 16 + lrow] = f2bf(S1[r]);
        }
        asm volatile("s_wait_dscnt 0" ::: "memory");  // lockstep wave32 visibility
        BFfrag pa;
#pragma unroll
        for (int j = 0; j < 8; ++j) {
            int mm0 = hi * 16 + 2 * j;
            pa.u[j] = *(const unsigned int*)(myP + lrow * LDSPAD + mm0);
        }

        // O += P * V^T over all 16 channel tiles (K-dim = 32 keys).
#pragma unroll
        for (int ct = 0; ct < 16; ++ct) {
            BFfrag bvf;
#pragma unroll
            for (int j = 0; j < 8; ++j) {
                int mm0 = hi * 16 + 2 * j;
                bvf.u[j] = *(const unsigned int*)(&ldsV[buf][0] + (ct * 16 + lrow) * LDSPAD + mm0);
            }
            O[ct] = __builtin_amdgcn_wmma_f32_16x16x32_bf16(false, pa.v, false, bvf.v,
                                                            (short)0, O[ct], false, false);
        }
    }

    // Epilogue: normalize, transpose each 16x16 tile through LDS, then
    // residual-add + store as coalesced b128.
    float rinv[8];
#pragma unroll
    for (int r = 0; r < 8; ++r) rinv[r] = 1.0f / rowsum[r];
    const float* xres = x + (size_t)b * CCH * NN;
    float* ob = out + (size_t)b * CCH * NN;
    const int c_row = lane >> 1, half = lane & 1;
#pragma unroll 1
    for (int ct = 0; ct < 16; ++ct) {
#pragma unroll
        for (int r = 0; r < 8; ++r)                    // scr[c][n], n padded to 20
            scrF[lrow * 20 + r + hi * 8] = O[ct][r] * rinv[r];
        asm volatile("s_wait_dscnt 0" ::: "memory");
#pragma unroll
        for (int q = 0; q < 2; ++q) {
            int nst = half * 8 + q * 4;
            v4f v = *(const v4f*)(scrF + c_row * 20 + nst);
            size_t idx = (size_t)(ct * 16 + c_row) * NN + qbase + nst;
            v4f xr = *(const v4f*)(xres + idx);
            *(v4f*)(ob + idx) = v + xr;
        }
        asm volatile("s_wait_dscnt 0" ::: "memory");   // reads done before reuse
    }
}

// ---------------------------------------------------------------------------

extern "C" void kernel_launch(void* const* d_in, const int* in_sizes, int n_in,
                              void* d_out, int out_size, void* d_ws, size_t ws_size,
                              hipStream_t stream) {
    (void)in_sizes; (void)n_in; (void)out_size; (void)ws_size;
    const float* x  = (const float*)d_in[0];
    const float* Wq = (const float*)d_in[1];
    const float* bq = (const float*)d_in[2];
    const float* Wk = (const float*)d_in[3];
    const float* bk = (const float*)d_in[4];
    const float* Wv = (const float*)d_in[5];
    const float* bv = (const float*)d_in[6];
    float* out = (float*)d_out;

    // Workspace (~38 MB): [ball f32 x512] [xT] [qT] [kT] [v] [Wall]
    float* ball = (float*)d_ws;
    unsigned short* base = (unsigned short*)((char*)d_ws + 512 * sizeof(float));
    const size_t XB = (size_t)BATCH * CCH * NN;   // 8,388,608
    const size_t QB = (size_t)BATCH * NN * DD;    // 1,048,576
    unsigned short* xT   = base;                  // [b][n][c]
    unsigned short* qT   = xT + XB;               // [b][n][d]
    unsigned short* kT   = qT + QB;               // [b][n][d]
    unsigned short* vb   = kT + QB;               // [b][c][n]
    unsigned short* Wall = vb + XB;               // [320][256]

    k_cvt_w <<<(320 * 256 + 255) / 256, 256, 0, stream>>>(Wq, bq, Wk, bk, Wv, bv, Wall, ball);
    k_cvt_xT<<<dim3(NN / 32, CCH / 32, BATCH), 256, 0, stream>>>(x, xT);
    k_projQK<<<dim3(NN / 16, 4, BATCH), 32, 0, stream>>>(xT, Wall, ball, qT, kT);
    k_projV <<<dim3(NN / 16, 16, BATCH), 32, 0, stream>>>(xT, Wall, ball, vb);
    k_attn  <<<dim3(NN / 128, BATCH), 256, 0, stream>>>(qT, kT, vb, x, out);
}
